// HospitalStaffingModel_67482526155021
// MI455X (gfx1250) — matrix operand order repaired and checked by reading, not compiled, not run on hardware
//
#include <hip/hip_runtime.h>

#define B_TOTAL 131072
#define D_IN 64
#define H1D 512
#define H2D 256
#define NCLS 128
#define EPSV 1e-5f
#define NEGV (-1e9f)

#define ROWS 64   // batch rows per block = 4 M-tiles

typedef __attribute__((ext_vector_type(16))) _Float16 v16h;
typedef __attribute__((ext_vector_type(8)))  _Float16 v8h;
typedef __attribute__((ext_vector_type(4)))  _Float16 v4h;
typedef __attribute__((ext_vector_type(8)))  float    v8f;

// LDS row pitches (in halfs), padded for bank-conflict avoidance, 16B-aligned rows
#define XP  72
#define H0P 520
#define H1P 264
#define HHP 72

#define WMMA_F16(a, b, c) \
  __builtin_amdgcn_wmma_f32_16x16x32_f16(false, (a), false, (b), (short)0, (c), false, false)

// ---------------- prep: fold BN scale/bias ----------------
__global__ void prep_scale_bias(const float* __restrict__ g0, const float* __restrict__ b0,
                                const float* __restrict__ be0, const float* __restrict__ rm0,
                                const float* __restrict__ rv0,
                                const float* __restrict__ g1, const float* __restrict__ b1,
                                const float* __restrict__ be1, const float* __restrict__ rm1,
                                const float* __restrict__ rv1,
                                float* __restrict__ s0, float* __restrict__ c0,
                                float* __restrict__ s1, float* __restrict__ c1) {
  int i = threadIdx.x;
  if (i < H1D) {
    float s = g0[i] * rsqrtf(rv0[i] + EPSV);
    s0[i] = s;
    c0[i] = (b0[i] - rm0[i]) * s + be0[i];
  }
  if (i < H2D) {
    float s = g1[i] * rsqrtf(rv1[i] + EPSV);
    s1[i] = s;
    c1[i] = (b1[i] - rm1[i]) * s + be1[i];
  }
}

// ---------------- prep: swizzle K x N f32 weight into WMMA B-fragment order ----------------
__global__ void prep_swizzle(const float* __restrict__ src, const float* __restrict__ scale,
                             _Float16* __restrict__ dst, int K, int N) {
  int t = threadIdx.x + blockIdx.x * blockDim.x;
  if (t >= K * N) return;
  int KT   = K >> 5;
  int tile = t >> 9;
  int w    = t & 511;
  int lane = w >> 4;
  int p    = w & 15;
  int nt   = tile / KT;
  int kt   = tile - nt * KT;
  int k    = kt * 32 + ((lane & 16) >> 1) + (p & 7) + ((p & 8) << 1);
  int n    = nt * 16 + (lane & 15);
  float v = src[k * N + n];
  if (scale) v *= scale[n];
  dst[t] = (_Float16)v;
}

// ---------------- fragment helpers ----------------
__device__ __forceinline__ v16h load_a(const _Float16* lds, int rowBase, int pitch, int kOff) {
  const int lane = threadIdx.x & 31;
  const int r  = lane & 15;
  const int kb = (lane & 16) ? 8 : 0;   // lanes 16..31 cover K-chunks +8 / +24
  const _Float16* p = lds + (rowBase + r) * pitch + kOff + kb;
  v8h lo = *(const v8h*)(p);
  v8h hi = *(const v8h*)(p + 16);
  v16h a;
#pragma unroll
  for (int i = 0; i < 8; ++i) { a[i] = lo[i]; a[i + 8] = hi[i]; }
  return a;
}

__device__ __forceinline__ v16h load_b(const _Float16* __restrict__ w, int tileIdx) {
  const int lane = threadIdx.x & 31;
  return *(const v16h*)(w + ((size_t)tileIdx * 32 + lane) * 16);
}

// Bias+ReLU epilogue: single base pointer, constant g*PITCH offsets so the
// backend folds per-row offsets into the ds_store immediate-offset field.
template <int PITCH>
__device__ __forceinline__ void store_c_relu(_Float16* lds, int rowBase, int colBase,
                                             v8f acc, const float* __restrict__ bias) {
  const int lane = threadIdx.x & 31;
  const int col  = colBase + (lane & 15);
  const int rsel = (lane & 16) ? 8 : 0;
  const float bv = bias[col];
  _Float16* p = lds + (rowBase + rsel) * PITCH + col;
#pragma unroll
  for (int g = 0; g < 8; ++g) {
    float v = acc[g] + bv;
    v = v > 0.f ? v : 0.f;
    p[g * PITCH] = (_Float16)v;
  }
}

// ---------------- fused MLP: 64 rows per block, 8 wave32s ----------------
// Each wave covers all FOUR 16-row M-tiles for its N slice: each B (weight)
// fragment feeds 4 WMMAs; weights stay L2-resident (pre-swizzled, ~426 KB).
__global__ __launch_bounds__(256) void fused_mlp(
    const float* __restrict__ x,
    const _Float16* __restrict__ W0h, const float* __restrict__ c0,
    const _Float16* __restrict__ W1h, const float* __restrict__ c1,
    const _Float16* __restrict__ nW1h, const float* __restrict__ nb1,
    const _Float16* __restrict__ nW2h, const float* __restrict__ nb2,
    const _Float16* __restrict__ dW1h, const float* __restrict__ db1,
    const _Float16* __restrict__ dW2h, const float* __restrict__ db2,
    float* __restrict__ out) {
  __shared__ _Float16 xs[ROWS * XP];        //   9216 B
  __shared__ _Float16 h0s[ROWS * H0P];      //  66560 B
  __shared__ _Float16 h1s[ROWS * H1P];      //  33792 B
  __shared__ _Float16 hhs[2][ROWS * HHP];   //  18432 B
  __shared__ int sthr[2][ROWS];             //    512 B   (total 128512 B)

  const int tid  = threadIdx.x;
  const int lane = tid & 31;
  const int wid  = tid >> 5;
  const int rowBase = blockIdx.x * ROWS;

  // stage in x tile as f16 (b128 global loads, b64 LDS stores), thresholds
  {
    const float4* xv = (const float4*)(x + (size_t)rowBase * D_IN);
#pragma unroll
    for (int i = tid; i < ROWS * 16; i += 256) {
      int r = i >> 4, c4 = i & 15;
      float4 v = xv[r * 16 + c4];
      v4h h; h[0] = (_Float16)v.x; h[1] = (_Float16)v.y;
             h[2] = (_Float16)v.z; h[3] = (_Float16)v.w;
      *(v4h*)&xs[r * XP + c4 * 4] = h;
    }
    if (tid < ROWS) {
      sthr[0][tid] = (int)x[(size_t)(rowBase + tid) * D_IN + (D_IN - 4)];
      sthr[1][tid] = (int)x[(size_t)(rowBase + tid) * D_IN + (D_IN - 3)];
    }
  }
  __syncthreads();

  // ---- layer 0: [64,64] @ [64,512] -> h0 ----
  {
    const int nb = wid * 4;
    v16h a[4][2];
#pragma unroll
    for (int mt = 0; mt < 4; ++mt)
#pragma unroll
      for (int kt = 0; kt < 2; ++kt)
        a[mt][kt] = load_a(xs, mt * 16, XP, kt * 32);
#pragma unroll
    for (int t = 0; t < 4; ++t) {
      const int n = nb + t;
      v16h b0 = load_b(W0h, n * 2 + 0);
      v16h b1 = load_b(W0h, n * 2 + 1);
#pragma unroll
      for (int mt = 0; mt < 4; ++mt) {
        v8f acc = {};
        acc = WMMA_F16(a[mt][0], b0, acc);
        acc = WMMA_F16(a[mt][1], b1, acc);
        store_c_relu<H0P>(h0s, mt * 16, n * 16, acc, c0);
      }
    }
  }
  __syncthreads();

  // ---- layer 1: [64,512] @ [512,256] -> h1 (4 M-tiles x 2 N per wave) ----
  {
    const int n0 = wid * 2, n1 = wid * 2 + 1;
    v8f acc[4][2] = {};
#pragma unroll
    for (int kt = 0; kt < 16; ++kt) {
      v16h b0 = load_b(W1h, n0 * 16 + kt);
      v16h b1 = load_b(W1h, n1 * 16 + kt);
#pragma unroll
      for (int mt = 0; mt < 4; ++mt) {
        v16h am = load_a(h0s, mt * 16, H0P, kt * 32);
        acc[mt][0] = WMMA_F16(am, b0, acc[mt][0]);
        acc[mt][1] = WMMA_F16(am, b1, acc[mt][1]);
      }
    }
#pragma unroll
    for (int mt = 0; mt < 4; ++mt) {
      store_c_relu<H1P>(h1s, mt * 16, n0 * 16, acc[mt][0], c1);
      store_c_relu<H1P>(h1s, mt * 16, n1 * 16, acc[mt][1], c1);
    }
  }
  __syncthreads();

  // ---- head hidden layers: 2 x ([64,256] @ [256,64]) ----
  {
    const int head = wid & 1;
    const int n    = wid >> 1;  // 0..3
    const _Float16* wgt = head ? dW1h : nW1h;
    const float*   bias = head ? db1 : nb1;
    v8f acc[4] = {};
#pragma unroll
    for (int kt = 0; kt < 8; ++kt) {
      v16h b = load_b(wgt, n * 8 + kt);
#pragma unroll
      for (int mt = 0; mt < 4; ++mt) {
        v16h am = load_a(h1s, mt * 16, H1P, kt * 32);
        acc[mt] = WMMA_F16(am, b, acc[mt]);
      }
    }
#pragma unroll
    for (int mt = 0; mt < 4; ++mt)
      store_c_relu<HHP>(hhs[head], mt * 16, n * 16, acc[mt], bias);
  }
  __syncthreads();

  // ---- head output layers + range mask: 2 x ([64,64] @ [64,128]) ----
  {
    const int head = wid & 1;
    const _Float16* wgt = head ? dW2h : nW2h;
    const float*   bias = head ? db2 : nb2;
    v16h a[4][2];
#pragma unroll
    for (int mt = 0; mt < 4; ++mt)
#pragma unroll
      for (int kt = 0; kt < 2; ++kt)
        a[mt][kt] = load_a(hhs[head], mt * 16, HHP, kt * 32);
    const int rsel = (lane & 16) ? 8 : 0;
#pragma unroll
    for (int u = 0; u < 2; ++u) {
      const int n = (wid >> 1) + u * 4;  // 0..7
      v16h b0 = load_b(wgt, n * 2 + 0);
      v16h b1 = load_b(wgt, n * 2 + 1);
      const int col  = n * 16 + (lane & 15);
      const float bv = bias[col];
#pragma unroll
      for (int mt = 0; mt < 4; ++mt) {
        v8f acc = {};
        acc = WMMA_F16(a[mt][0], b0, acc);
        acc = WMMA_F16(a[mt][1], b1, acc);
        // single base pointer; g*NCLS*4B folds into store immediate offsets
        float* po = out + (size_t)head * B_TOTAL * NCLS
                        + (size_t)(rowBase + mt * 16 + rsel) * NCLS + col;
        const int* th = &sthr[head][mt * 16 + rsel];
#pragma unroll
        for (int g = 0; g < 8; ++g) {
          float v = acc[g] + bv;
          v = (col <= th[g]) ? v : NEGV;
          po[g * NCLS] = v;
        }
      }
    }
  }
}

extern "C" void kernel_launch(void* const* d_in, const int* in_sizes, int n_in,
                              void* d_out, int out_size, void* d_ws, size_t ws_size,
                              hipStream_t stream) {
  const float* x   = (const float*)d_in[0];
  const float* W0  = (const float*)d_in[1];
  const float* b0  = (const float*)d_in[2];
  const float* g0  = (const float*)d_in[3];
  const float* be0 = (const float*)d_in[4];
  const float* rm0 = (const float*)d_in[5];
  const float* rv0 = (const float*)d_in[6];
  const float* W1  = (const float*)d_in[7];
  const float* b1  = (const float*)d_in[8];
  const float* g1  = (const float*)d_in[9];
  const float* be1 = (const float*)d_in[10];
  const float* rm1 = (const float*)d_in[11];
  const float* rv1 = (const float*)d_in[12];
  const float* nW1 = (const float*)d_in[13];
  const float* nb1 = (const float*)d_in[14];
  const float* nW2 = (const float*)d_in[15];
  const float* nb2 = (const float*)d_in[16];
  const float* dW1 = (const float*)d_in[17];
  const float* db1 = (const float*)d_in[18];
  const float* dW2 = (const float*)d_in[19];
  const float* db2 = (const float*)d_in[20];

  char* ws = (char*)d_ws;
  _Float16* W0h  = (_Float16*)(ws + 0);        // 64*512  f16 = 65536 B
  _Float16* W1h  = (_Float16*)(ws + 65536);    // 512*256 f16 = 262144 B
  _Float16* nW1h = (_Float16*)(ws + 327680);   // 256*64  f16 = 32768 B
  _Float16* dW1h = (_Float16*)(ws + 360448);   // 256*64  f16 = 32768 B
  _Float16* nW2h = (_Float16*)(ws + 393216);   // 64*128  f16 = 16384 B
  _Float16* dW2h = (_Float16*)(ws + 409600);   // 64*128  f16 = 16384 B
  float*    c0w  = (float*)(ws + 425984);      // 512 f32
  float*    c1w  = (float*)(ws + 428032);      // 256 f32
  float*    s0w  = (float*)(ws + 429056);      // 512 f32
  float*    s1w  = (float*)(ws + 431104);      // 256 f32

  prep_scale_bias<<<1, 512, 0, stream>>>(g0, b0, be0, rm0, rv0,
                                         g1, b1, be1, rm1, rv1,
                                         s0w, c0w, s1w, c1w);

  prep_swizzle<<<(64 * 512 + 255) / 256, 256, 0, stream>>>(W0, s0w, W0h, 64, 512);
  prep_swizzle<<<(512 * 256 + 255) / 256, 256, 0, stream>>>(W1, s1w, W1h, 512, 256);
  prep_swizzle<<<(256 * 64 + 255) / 256, 256, 0, stream>>>(nW1, nullptr, nW1h, 256, 64);
  prep_swizzle<<<(256 * 64 + 255) / 256, 256, 0, stream>>>(dW1, nullptr, dW1h, 256, 64);
  prep_swizzle<<<(64 * 128 + 255) / 256, 256, 0, stream>>>(nW2, nullptr, nW2h, 64, 128);
  prep_swizzle<<<(64 * 128 + 255) / 256, 256, 0, stream>>>(dW2, nullptr, dW2h, 64, 128);

  fused_mlp<<<B_TOTAL / ROWS, 256, 0, stream>>>(x,
                                                W0h, c0w, W1h, c1w,
                                                nW1h, nb1, nW2h, nb2,
                                                dW1h, db1, dW2h, db2,
                                                (float*)d_out);
}
